// GnnEncoder_39848706573800
// MI455X (gfx1250) — matrix operand order repaired and checked by reading, not compiled, not run on hardware
//
#include <hip/hip_runtime.h>
#include <hip/hip_bf16.h>

// ---------------------------------------------------------------------------
// GNN encoder for MI455X (gfx1250, wave32, WMMA f32_16x16x32_f16).
// Shapes: N=50000 nodes, E=600000 edges, C=128, d=129 (pad K->160, cols->144).
// ---------------------------------------------------------------------------

#define D_REAL 129          // real feature dim (C+1)
#define DP     132          // f32 row stride (pad to multiple of 4)
#define KP     160          // padded K for WMMA (5 tiles of 32)
#define NT     9            // col tiles (144 cols)
#define CC     128          // C
#define WAVES  2            // waves per block in WMMA kernels
#define WBLK   (NT*16*KP)   // halves per packed weight block (144*160)

typedef _Float16 v16h __attribute__((ext_vector_type(16)));
typedef _Float16 v8h  __attribute__((ext_vector_type(8)));
typedef float    v8f  __attribute__((ext_vector_type(8)));

// A-fragment: 16x32 f16 tile, row m = lane&15, lane<16 -> K 0..15 contiguous, lane>=16 -> 16..31
__device__ __forceinline__ v16h load_frag_a(const _Float16* p) {
  v8h lo = *(const v8h*)(p);
  v8h hi = *(const v8h*)(p + 8);
  return __builtin_shufflevector(lo, hi, 0,1,2,3,4,5,6,7,8,9,10,11,12,13,14,15);
}
// B-fragment: Wt stored [n][k]; lane<16 -> K 0..15 contiguous, lane>=16 -> 16..31
__device__ __forceinline__ v16h load_frag_b(const _Float16* p) {
  v8h lo = *(const v8h*)(p);
  v8h hi = *(const v8h*)(p + 8);
  return __builtin_shufflevector(lo, hi, 0,1,2,3,4,5,6,7,8,9,10,11,12,13,14,15);
}

__device__ __forceinline__ v8f vzero8() {
  v8f z = {0.f,0.f,0.f,0.f,0.f,0.f,0.f,0.f};
  return z;
}

// --------------------------- weight packing --------------------------------
// dst[n*KP+k] = (f16) W[k*wCols+n], zero padded to 144x160.
__global__ void pack_weight_kernel(const float* __restrict__ W, int wCols, int nReal,
                                   _Float16* __restrict__ dst) {
  int i = blockIdx.x * blockDim.x + threadIdx.x;
  if (i >= WBLK) return;
  int n = i / KP, k = i - n * KP;
  float v = 0.f;
  if (n < nReal && k < D_REAL) v = W[k * wCols + n];
  dst[i] = (_Float16)v;
}

// --------------------------- xc build --------------------------------------
__global__ void build_xc0_kernel(const float* __restrict__ x,
                                 const float* __restrict__ isc,
                                 float* __restrict__ xc, int N) {
  int i = blockIdx.x * blockDim.x + threadIdx.x;
  if (i >= N * DP) return;
  int r = i / DP, j = i - r * DP;
  float v = 0.f;
  if (j < CC) v = x[r * CC + j];
  else if (j == CC) v = isc[r];
  xc[i] = v;
}

// h = xc + agg (cols < 129), pad zeros
__global__ void add_h_kernel(const float* __restrict__ a, const float* __restrict__ b,
                             float* __restrict__ o, int N) {
  int i = blockIdx.x * blockDim.x + threadIdx.x;
  if (i >= N * DP) return;
  int j = i % DP;
  o[i] = (j < D_REAL) ? (a[i] + b[i]) : 0.f;
}

// --------------------- fused edge-MLP + message + scatter ------------------
// Per 16-edge tile: A = relu(ea*l0w + l0b) (f16), e = A @ l1w (WMMA),
// msg = relu(xc[src] + e + l1b), agg[dst] += msg (global f32 atomics).
__global__ void __launch_bounds__(64)
msg_kernel(const float* __restrict__ ea, const float* __restrict__ l0w,
           const float* __restrict__ l0b, const float* __restrict__ l1b,
           const _Float16* __restrict__ WtE, const float* __restrict__ xc,
           const int* __restrict__ eidx, float* __restrict__ agg,
           int E, int ntilesE) {
  __shared__ _Float16 sB[WBLK];
  __shared__ _Float16 sA[WAVES][16 * KP];
  __shared__ float sL0W[KP], sL0B[KP], sL1B[NT * 16];
  __shared__ float sEA[WAVES][16];
  __shared__ int   sSrc[WAVES][16], sDst[WAVES][16];

  const int tid = threadIdx.x;
  for (int i = tid; i < WBLK / 8; i += 64)
    ((int4*)sB)[i] = ((const int4*)WtE)[i];
  for (int i = tid; i < KP; i += 64) {
    sL0W[i] = (i < D_REAL) ? l0w[i] : 0.f;
    sL0B[i] = (i < D_REAL) ? l0b[i] : 0.f;
  }
  for (int i = tid; i < NT * 16; i += 64)
    sL1B[i] = (i < D_REAL) ? l1b[i] : 0.f;
  __syncthreads();

  const int wv = tid >> 5, lane = tid & 31;
  const int laneM = lane & 15, hi = lane >> 4;

  for (int t = blockIdx.x * WAVES + wv; t < ntilesE; t += gridDim.x * WAVES) {
    const int e0 = t * 16;
    if (lane < 16) {
      sEA[wv][lane]  = ea[e0 + lane];
      sSrc[wv][lane] = eidx[e0 + lane];
      sDst[wv][lane] = eidx[E + e0 + lane];
    }
    // build A tile [16][KP] f16
    for (int i = lane; i < 16 * KP; i += 32) {
      int m = i / KP, k = i - m * KP;
      float v = fmaxf(sEA[wv][m] * sL0W[k] + sL0B[k], 0.f);
      if (k >= D_REAL) v = 0.f;
      sA[wv][i] = (_Float16)v;
    }
    v8f acc[NT];
#pragma unroll
    for (int n = 0; n < NT; ++n) acc[n] = vzero8();
#pragma unroll
    for (int kt = 0; kt < 5; ++kt) {
      const v16h af = load_frag_a(&sA[wv][laneM * KP + kt * 32 + hi * 16]);
#pragma unroll
      for (int n = 0; n < NT; ++n) {
        const v16h bf = load_frag_b(&sB[(n * 16 + laneM) * KP + kt * 32 + hi * 16]);
        acc[n] = __builtin_amdgcn_wmma_f32_16x16x32_f16(
            false, af, false, bf, (short)0, acc[n], false, false);
      }
    }
    // epilogue: relu(xc[src] + e) scattered into agg[dst]
    const int mbase = hi * 8;
#pragma unroll
    for (int n = 0; n < NT; ++n) {
      const int col = n * 16 + laneM;
      if (col >= D_REAL) continue;
      const float bias = sL1B[col];
#pragma unroll
      for (int r = 0; r < 8; ++r) {
        const int m = mbase + r;
        float msg = xc[(long)sSrc[wv][m] * DP + col] + acc[n][r] + bias;
        if (msg > 0.f)
          unsafeAtomicAdd(&agg[(long)sDst[wv][m] * DP + col], msg);
      }
    }
  }
}

// --------------------------- node GEMM (WMMA) ------------------------------
// out[M x ncols] = A[rows x 129] @ Wt + bias (+ resid). Optional row gather.
__global__ void __launch_bounds__(64)
gemm_kernel(const float* __restrict__ A, const int* __restrict__ gather, int M,
            const _Float16* __restrict__ Wt, const float* __restrict__ bias,
            int ncols, float* __restrict__ out, int outStride,
            const float* __restrict__ resid, int ntilesRow) {
  __shared__ _Float16 sB[WBLK];
  __shared__ _Float16 sA[WAVES][16 * KP];
  __shared__ float sBias[NT * 16];

  const int tid = threadIdx.x;
  for (int i = tid; i < WBLK / 8; i += 64)
    ((int4*)sB)[i] = ((const int4*)Wt)[i];
  for (int i = tid; i < NT * 16; i += 64)
    sBias[i] = (i < ncols) ? bias[i] : 0.f;
  __syncthreads();

  const int wv = tid >> 5, lane = tid & 31;
  const int laneM = lane & 15, hi = lane >> 4;

  for (int t = blockIdx.x * WAVES + wv; t < ntilesRow; t += gridDim.x * WAVES) {
    for (int i = lane; i < 16 * KP; i += 32) {
      int m = i / KP, k = i - m * KP;
      int row = t * 16 + m;
      float v = 0.f;
      if (k < D_REAL && row < M) {
        int g = gather ? gather[row] : row;
        v = A[(long)g * DP + k];
      }
      sA[wv][i] = (_Float16)v;
    }
    v8f acc[NT];
#pragma unroll
    for (int n = 0; n < NT; ++n) acc[n] = vzero8();
#pragma unroll
    for (int kt = 0; kt < 5; ++kt) {
      const v16h af = load_frag_a(&sA[wv][laneM * KP + kt * 32 + hi * 16]);
#pragma unroll
      for (int n = 0; n < NT; ++n) {
        const v16h bf = load_frag_b(&sB[(n * 16 + laneM) * KP + kt * 32 + hi * 16]);
        acc[n] = __builtin_amdgcn_wmma_f32_16x16x32_f16(
            false, af, false, bf, (short)0, acc[n], false, false);
      }
    }
    const int mbase = hi * 8;
#pragma unroll
    for (int n = 0; n < NT; ++n) {
      const int col = n * 16 + laneM;
      if (col >= ncols) continue;
#pragma unroll
      for (int r = 0; r < 8; ++r) {
        const int row = t * 16 + mbase + r;
        if (row >= M) continue;
        float v = acc[n][r] + sBias[col];
        if (resid) {
          int g = gather ? gather[row] : row;
          v += resid[(long)g * CC + col];
        }
        out[(long)row * outStride + col] = v;
      }
    }
  }
}

// --------------------------- batch-norm helpers ----------------------------
__global__ void __launch_bounds__(160)
bn_stats_kernel(const float* __restrict__ in, int stride, int ncols, int rows,
                int chunk, float* __restrict__ stats) {
  const int j = threadIdx.x;
  if (j >= ncols) return;
  const long r0 = (long)blockIdx.x * chunk;
  float s = 0.f, q = 0.f;
  for (int r = 0; r < chunk; ++r) {
    long row = r0 + r;
    if (row >= rows) break;
    float v = in[row * stride + j];
    s += v;
    q += v * v;
  }
  unsafeAtomicAdd(&stats[j], s);
  unsafeAtomicAdd(&stats[144 + j], q);
}

__global__ void __launch_bounds__(160)
bn_finalize_kernel(const float* __restrict__ g, const float* __restrict__ be,
                   int ncols, float invN, float* __restrict__ stats) {
  const int j = threadIdx.x;
  if (j >= ncols) return;
  float mu  = stats[j] * invN;
  float var = stats[144 + j] * invN - mu * mu;
  float sc  = g[j] * rsqrtf(var + 1e-5f);
  stats[288 + j] = sc;
  stats[432 + j] = be[j] - mu * sc;
}

__global__ void bn_relu_kernel(const float* __restrict__ in,
                               const float* __restrict__ stats,
                               float* __restrict__ out, int N) {
  int i = blockIdx.x * blockDim.x + threadIdx.x;
  if (i >= N * DP) return;
  int j = i % DP;
  float v = 0.f;
  if (j < D_REAL) v = fmaxf(in[i] * stats[288 + j] + stats[432 + j], 0.f);
  out[i] = v;
}

// xc2 = concat(relu(bn0(g1)), is_center), padded
__global__ void build_xc2_kernel(const float* __restrict__ g1,
                                 const float* __restrict__ stats,
                                 const float* __restrict__ isc,
                                 float* __restrict__ xc, int N) {
  int i = blockIdx.x * blockDim.x + threadIdx.x;
  if (i >= N * DP) return;
  int r = i / DP, j = i - r * DP;
  float v = 0.f;
  if (j < CC) v = fmaxf(g1[r * CC + j] * stats[288 + j] + stats[432 + j], 0.f);
  else if (j == CC) v = isc[r];
  xc[i] = v;
}

// ---------------------------------------------------------------------------
extern "C" void kernel_launch(void* const* d_in, const int* in_sizes, int n_in,
                              void* d_out, int out_size, void* d_ws, size_t ws_size,
                              hipStream_t stream) {
  const float* x     = (const float*)d_in[0];
  const float* eattr = (const float*)d_in[1];
  const float* isc   = (const float*)d_in[2];
  const float* l0_w  = (const float*)d_in[3];
  const float* l0_b  = (const float*)d_in[4];
  const float* l1_w  = (const float*)d_in[5];
  const float* l1_b  = (const float*)d_in[6];
  const float* a_w1  = (const float*)d_in[7];
  const float* a_b1  = (const float*)d_in[8];
  const float* a_g   = (const float*)d_in[9];
  const float* a_be  = (const float*)d_in[10];
  const float* a_w2  = (const float*)d_in[11];
  const float* a_b2  = (const float*)d_in[12];
  const float* b_w1  = (const float*)d_in[13];
  const float* b_b1  = (const float*)d_in[14];
  const float* b_g   = (const float*)d_in[15];
  const float* b_be  = (const float*)d_in[16];
  const float* b_w2  = (const float*)d_in[17];
  const float* b_b2  = (const float*)d_in[18];
  const float* bn0_g = (const float*)d_in[19];
  const float* bn0_b = (const float*)d_in[20];
  const int*   eidx  = (const int*)d_in[21];
  const int*   cidx  = (const int*)d_in[22];
  float*       out   = (float*)d_out;

  const int N  = in_sizes[2];       // is_center has N elements
  const int E  = in_sizes[1];       // edge_attr has E elements
  const int NC = in_sizes[22];

  char* ws = (char*)d_ws;
  _Float16* W16   = (_Float16*)ws;                 // 5 * WBLK halves (~230 KB)
  float*    stats = (float*)(ws + (256 << 10));    // 576 floats
  float*    P0    = (float*)(ws + (1 << 20));      // xc
  const size_t SZ = (size_t)N * DP;
  float* P1 = P0 + SZ;                             // agg / z
  float* P2 = P1 + SZ;                             // h / zr
  float* P3 = P2 + SZ;                             // gine0 output [N,128]

  const int packGrid = (WBLK + 255) / 256;
  pack_weight_kernel<<<packGrid, 256, 0, stream>>>(l1_w, 129, 129, W16 + 0 * WBLK);
  pack_weight_kernel<<<packGrid, 256, 0, stream>>>(a_w1, 129, 129, W16 + 1 * WBLK);
  pack_weight_kernel<<<packGrid, 256, 0, stream>>>(a_w2, 128, 128, W16 + 2 * WBLK);
  pack_weight_kernel<<<packGrid, 256, 0, stream>>>(b_w1, 129, 129, W16 + 3 * WBLK);
  pack_weight_kernel<<<packGrid, 256, 0, stream>>>(b_w2, 128, 128, W16 + 4 * WBLK);

  const int gridNDP = (N * DP + 255) / 256;
  build_xc0_kernel<<<gridNDP, 256, 0, stream>>>(x, isc, P0, N);

  const int tilesE = E / 16;
  const int tilesN = (N + 15) / 16;
  const int statBlocks = (N + 249) / 250;

  // ---------------- conv0 (a_* weights) ----------------
  hipMemsetAsync(P1, 0, SZ * sizeof(float), stream);
  msg_kernel<<<2048, 64, 0, stream>>>(eattr, l0_w, l0_b, l1_b, W16 + 0 * WBLK,
                                      P0, eidx, P1, E, tilesE);
  add_h_kernel<<<gridNDP, 256, 0, stream>>>(P0, P1, P2, N);
  gemm_kernel<<<1024, 64, 0, stream>>>(P2, nullptr, N, W16 + 1 * WBLK, a_b1,
                                       129, P1, DP, nullptr, tilesN);
  hipMemsetAsync(stats, 0, 2 * 144 * sizeof(float), stream);
  bn_stats_kernel<<<statBlocks, 160, 0, stream>>>(P1, DP, 129, N, 250, stats);
  bn_finalize_kernel<<<1, 160, 0, stream>>>(a_g, a_be, 129, 1.f / N, stats);
  bn_relu_kernel<<<gridNDP, 256, 0, stream>>>(P1, stats, P2, N);
  gemm_kernel<<<1024, 64, 0, stream>>>(P2, nullptr, N, W16 + 2 * WBLK, a_b2,
                                       128, P3, 128, x, tilesN);

  // ---------------- bn0 + relu + concat ----------------
  hipMemsetAsync(stats, 0, 2 * 144 * sizeof(float), stream);
  bn_stats_kernel<<<statBlocks, 160, 0, stream>>>(P3, 128, 128, N, 250, stats);
  bn_finalize_kernel<<<1, 160, 0, stream>>>(bn0_g, bn0_b, 128, 1.f / N, stats);
  build_xc2_kernel<<<gridNDP, 256, 0, stream>>>(P3, stats, isc, P0, N);

  // ---------------- conv1 (b_* weights) ----------------
  hipMemsetAsync(P1, 0, SZ * sizeof(float), stream);
  msg_kernel<<<2048, 64, 0, stream>>>(eattr, l0_w, l0_b, l1_b, W16 + 0 * WBLK,
                                      P0, eidx, P1, E, tilesE);
  add_h_kernel<<<gridNDP, 256, 0, stream>>>(P0, P1, P2, N);
  gemm_kernel<<<1024, 64, 0, stream>>>(P2, nullptr, N, W16 + 3 * WBLK, b_b1,
                                       129, P1, DP, nullptr, tilesN);
  hipMemsetAsync(stats, 0, 2 * 144 * sizeof(float), stream);
  bn_stats_kernel<<<statBlocks, 160, 0, stream>>>(P1, DP, 129, N, 250, stats);
  bn_finalize_kernel<<<1, 160, 0, stream>>>(b_g, b_be, 129, 1.f / N, stats);
  bn_relu_kernel<<<gridNDP, 256, 0, stream>>>(P1, stats, P2, N);

  // final GEMM only on center rows, writes d_out with +x residual
  const int tilesC = (NC + 15) / 16;
  gemm_kernel<<<(tilesC + WAVES - 1) / WAVES, 64, 0, stream>>>(
      P2, cidx, NC, W16 + 4 * WBLK, b_b2, 128, out, 128, x, tilesC);
}
